// MultiHeadAttentionBlock_3590592659731
// MI455X (gfx1250) — compile-verified, matrix operand order checked by reading, not run on hardware
//
#include <hip/hip_runtime.h>
#include <hip/hip_bf16.h>
#include <cstddef>
#include <cstdint>

typedef __attribute__((ext_vector_type(16))) __bf16 v16bf;
typedef __attribute__((ext_vector_type(8)))  __bf16 v8bf;
typedef __attribute__((ext_vector_type(8)))  float  v8f;
typedef unsigned int v4u __attribute__((ext_vector_type(4)));
typedef int          v8i __attribute__((ext_vector_type(8)));
typedef int          v4i __attribute__((ext_vector_type(4)));

#define WMMA_BF16(a,b,c) \
  __builtin_amdgcn_wmma_f32_16x16x32_bf16(false,(a),false,(b),(short)0,(c),false,false)

constexpr int Bb = 4, Ss = 2048, Ee = 1024, Hh = 16, Dh = 64;
constexpr int Mm = Bb * Ss; // 8192

#if __has_builtin(__builtin_amdgcn_tensor_load_to_lds)
#define USE_TDM 1
#else
#define USE_TDM 0
#endif

// ---------------- f32 -> bf16 conversion pre-pass ----------------
__global__ __launch_bounds__(256)
void cvt_f32_bf16_kernel(const float* __restrict__ in, __bf16* __restrict__ out, int n)
{
  const int i = (blockIdx.x * 256 + threadIdx.x) * 8;
  if (i < n) {
    float4 a = *(const float4*)(in + i);
    float4 b = *(const float4*)(in + i + 4);
    v8bf o;
    o[0] = (__bf16)a.x; o[1] = (__bf16)a.y; o[2] = (__bf16)a.z; o[3] = (__bf16)a.w;
    o[4] = (__bf16)b.x; o[5] = (__bf16)b.y; o[6] = (__bf16)b.z; o[7] = (__bf16)b.w;
    *(v8bf*)(out + i) = o;
  }
}

#if USE_TDM
__device__ __forceinline__ uint32_t lds_addr_of(const void* p)
{
  // generic pointer to LDS: addr[31:0] is the LDS byte offset (ISA aperture rule)
  return (uint32_t)(uint64_t)(uintptr_t)p;
}

// TDM 2D tile load: tile = 32 (contiguous, bf16) x 128 rows, row stride = Ee.
// D# per CDNA5 ISA: group0 {count=1, lds_addr, global_addr,
// type=2}, group1 {data_size=1(2B), tensor_dim0=Ee, tensor_dim1=rows,
// tile_dim0=32, tile_dim1=128, tensor_dim0_stride=Ee}. Groups 2/3 zero (2D).
__device__ __forceinline__ void tdm_load_tile(const __bf16* gsrc, uint32_t lds_off,
                                              uint32_t tensor_rows)
{
  const uint64_t ga = (uint64_t)(uintptr_t)gsrc;
  v4u g0 = { 1u,                                    // count = 1 valid descriptor
             lds_off,                               // LDS byte address
             (uint32_t)ga,                          // global_addr[31:0]
             (((uint32_t)(ga >> 32)) & 0x01FFFFFFu) // global_addr[56:32]
               | 0x80000000u };                     // type = 2 ("image")
  v8i g1 = { (int)0x00010000u,                      // data_size=1 (2 bytes)
             (int)((uint32_t)Ee << 16),             // tensor_dim0[15:0] @bits63:48
             (int)(tensor_rows << 16),              // tensor_dim1[15:0] @bits95:80
             (int)(32u << 16),                      // tile_dim0=32 @bits127:112
             128,                                   // tile_dim1=128
             Ee,                                    // tensor_dim0_stride[31:0]
             0, 0 };
  v4i gz4 = { 0, 0, 0, 0 };
  v8i gz8 = { 0, 0, 0, 0, 0, 0, 0, 0 };
  // 6-arg variant (clang-23 / therock-10.0 headers): extra int32x8 group before cpol
  __builtin_amdgcn_tensor_load_to_lds(g0, g1, gz4, gz4, gz8, 0);
}
#endif

// ---------------- bf16 GEMM: Out = A[M,K] x W[N,K]^T + bias ----------------
// OUT_MODE: 0 -> bf16 [B,H,S,DH] (Q,K)   1 -> bf16 [B,H,DH,S] (V, transposed)
//           2 -> f32  [M,E] flat (final projection)
template<int OUT_MODE>
__global__ __launch_bounds__(256)
void gemm_bf16_kernel(const __bf16* __restrict__ A,
                      const __bf16* __restrict__ W,
                      const float* __restrict__ bias,
                      void* __restrict__ Out)
{
  __shared__ __bf16 As[2][128 * 32];
  __shared__ __bf16 Bs[2][128 * 32];
  const int tid  = threadIdx.x;
  const int wave = tid >> 5, lane = tid & 31;
  const int wm = wave >> 1, wn = wave & 1;       // 4x2 wave grid
  const int half = lane >> 4, l16 = lane & 15;
  const int mblock = blockIdx.x * 128;
  const int nblock = blockIdx.y * 128;

  v8f acc[2][4] = {};

#if USE_TDM
  // Prime the pipeline: TDM fills buffer 0 while waves set up.
  if (wave == 0) {
    tdm_load_tile(A + (size_t)mblock * Ee, lds_addr_of(As[0]), Mm);
    tdm_load_tile(W + (size_t)nblock * Ee, lds_addr_of(Bs[0]), Ee);
  }
  int buf = 0;
  for (int k0 = 0; k0 < Ee; k0 += 32) {
    if (wave == 0) __builtin_amdgcn_s_wait_tensorcnt(0);
    __syncthreads();  // buffer `buf` ready for all; all waves done with buf^1
    if (wave == 0 && k0 + 32 < Ee) {
      tdm_load_tile(A + (size_t)mblock * Ee + k0 + 32, lds_addr_of(As[buf ^ 1]), Mm);
      tdm_load_tile(W + (size_t)nblock * Ee + k0 + 32, lds_addr_of(Bs[buf ^ 1]), Ee);
    }
#else
  const int ar = tid >> 1;         // 0..127
  const int ac = (tid & 1) * 16;   // 0 or 16
  int buf = 0;
  for (int k0 = 0; k0 < Ee; k0 += 32) {
    {
      const __bf16* ap = A + (size_t)(mblock + ar) * Ee + k0 + ac;
      __bf16* d = &As[0][ar * 32 + ac];
      *(v8bf*)d       = *(const v8bf*)ap;
      *(v8bf*)(d + 8) = *(const v8bf*)(ap + 8);
      const __bf16* wp = W + (size_t)(nblock + ar) * Ee + k0 + ac;
      __bf16* e = &Bs[0][ar * 32 + ac];
      *(v8bf*)e       = *(const v8bf*)wp;
      *(v8bf*)(e + 8) = *(const v8bf*)(wp + 8);
    }
    __syncthreads();
#endif

    // ---- fragments: per-lane two 16B chunks matching 16-bit A-operand layout ----
    v16bf af[2], bfr[4];
    #pragma unroll
    for (int mt = 0; mt < 2; ++mt) {
      const __bf16* p = &As[buf][(wm*32 + mt*16 + l16) * 32];
      ((v8bf*)&af[mt])[0] = *(const v8bf*)(p + half*8);
      ((v8bf*)&af[mt])[1] = *(const v8bf*)(p + 16 + half*8);
    }
    #pragma unroll
    for (int nt = 0; nt < 4; ++nt) {
      const __bf16* p = &Bs[buf][(wn*64 + nt*16 + l16) * 32];
      ((v8bf*)&bfr[nt])[0] = *(const v8bf*)(p + half*8);
      ((v8bf*)&bfr[nt])[1] = *(const v8bf*)(p + 16 + half*8);
    }
    #pragma unroll
    for (int mt = 0; mt < 2; ++mt)
      #pragma unroll
      for (int nt = 0; nt < 4; ++nt)
        acc[mt][nt] = WMMA_BF16(af[mt], bfr[nt], acc[mt][nt]);

#if USE_TDM
    buf ^= 1;
#else
    __syncthreads();
#endif
  }

  // ---- epilogue: C/D layout M = r + 8*half, N = l16 ----
  #pragma unroll
  for (int mt = 0; mt < 2; ++mt) {
    #pragma unroll
    for (int nt = 0; nt < 4; ++nt) {
      const int Ni = nblock + wn*64 + nt*16 + l16;
      const float bvl = bias[Ni];
      #pragma unroll
      for (int r = 0; r < 8; ++r) {
        const int Mi = mblock + wm*32 + mt*16 + r + 8*half;
        const float val = acc[mt][nt][r] + bvl;
        if constexpr (OUT_MODE == 2) {
          ((float*)Out)[(size_t)Mi * Ee + Ni] = val;
        } else {
          const int b = Mi >> 11, s = Mi & (Ss - 1);
          const int h = Ni >> 6,  d = Ni & (Dh - 1);
          if constexpr (OUT_MODE == 0)
            ((__bf16*)Out)[(((size_t)b*Hh + h)*Ss + s)*Dh + d] = (__bf16)val;
          else
            ((__bf16*)Out)[(((size_t)b*Hh + h)*Dh + d)*Ss + s] = (__bf16)val;
        }
      }
    }
  }
}

// ---------------- Flash attention (causal), per-wave 16-row Q strip ----------------
__global__ __launch_bounds__(256)
void attn_kernel(const __bf16* __restrict__ Qh,   // [B,H,S,DH]
                 const __bf16* __restrict__ Kh,   // [B,H,S,DH]
                 const __bf16* __restrict__ Vh,   // [B,H,DH,S]
                 __bf16* __restrict__ X)          // [B,S,E]
{
  __shared__ __bf16 Ps[8 * 16 * 32];  // per-wave 16x32 P staging
  const int tid = threadIdx.x;
  const int wave = tid >> 5, lane = tid & 31;
  const int half = lane >> 4, l16 = lane & 15;
  const int bh = blockIdx.y;
  const int qbase = blockIdx.x * 128 + wave * 16;

  const __bf16* qp = Qh + ((size_t)bh * Ss + qbase) * Dh;
  const __bf16* kp = Kh + (size_t)bh * Ss * Dh;
  const __bf16* vp = Vh + (size_t)bh * Dh * Ss;

  v16bf qf[2];
  #pragma unroll
  for (int c = 0; c < 2; ++c) {
    const __bf16* p = qp + (size_t)l16 * Dh + c*32 + half*8;
    ((v8bf*)&qf[c])[0] = *(const v8bf*)p;
    ((v8bf*)&qf[c])[1] = *(const v8bf*)(p + 16);
  }

  float mrow[8], lrow[8];
  v8f oacc[4] = {};
  #pragma unroll
  for (int r = 0; r < 8; ++r) { mrow[r] = -1e30f; lrow[r] = 0.f; }

  __bf16* pw = &Ps[wave * 512];
  const float scale = 0.125f;  // 1/sqrt(DH)

  const int kv_hi = qbase + 16;  // causal bound for this wave's strip
  for (int kvb = 0; kvb < kv_hi; kvb += 32) {
    v8f sc[2];
    #pragma unroll
    for (int jj = 0; jj < 2; ++jj) {
      v16bf kf0, kf1;
      const __bf16* p = kp + (size_t)(kvb + jj*16 + l16) * Dh + half*8;
      ((v8bf*)&kf0)[0] = *(const v8bf*)p;
      ((v8bf*)&kf0)[1] = *(const v8bf*)(p + 16);
      ((v8bf*)&kf1)[0] = *(const v8bf*)(p + 32);
      ((v8bf*)&kf1)[1] = *(const v8bf*)(p + 48);
      v8f s = {};
      s = WMMA_BF16(qf[0], kf0, s);
      s = WMMA_BF16(qf[1], kf1, s);
      sc[jj] = s;
    }
    #pragma unroll
    for (int jj = 0; jj < 2; ++jj) {
      const int kvcol = kvb + jj*16 + l16;
      #pragma unroll
      for (int r = 0; r < 8; ++r) {
        const int qrow = qbase + r + 8*half;
        sc[jj][r] = (kvcol > qrow) ? -1e30f : sc[jj][r] * scale;
      }
    }
    float p0[8], p1[8];
    #pragma unroll
    for (int r = 0; r < 8; ++r) {
      float mx = fmaxf(sc[0][r], sc[1][r]);
      #pragma unroll
      for (int off = 8; off; off >>= 1)
        mx = fmaxf(mx, __shfl_xor(mx, off, 16));
      const float mnew  = fmaxf(mrow[r], mx);
      const float alpha = __expf(mrow[r] - mnew);
      mrow[r] = mnew;
      p0[r] = __expf(sc[0][r] - mnew);
      p1[r] = __expf(sc[1][r] - mnew);
      float rs = p0[r] + p1[r];
      #pragma unroll
      for (int off = 8; off; off >>= 1)
        rs += __shfl_xor(rs, off, 16);
      lrow[r] = lrow[r] * alpha + rs;
      #pragma unroll
      for (int nt = 0; nt < 4; ++nt) oacc[nt][r] *= alpha;
    }
    #pragma unroll
    for (int r = 0; r < 8; ++r) {
      pw[(r + 8*half)*32 + l16]      = (__bf16)p0[r];
      pw[(r + 8*half)*32 + 16 + l16] = (__bf16)p1[r];
    }
    v16bf pf;
    {
      const __bf16* pr = &Ps[wave*512 + l16*32];
      ((v8bf*)&pf)[0] = *(const v8bf*)(pr + half*8);
      ((v8bf*)&pf)[1] = *(const v8bf*)(pr + 16 + half*8);
    }
    #pragma unroll
    for (int nt = 0; nt < 4; ++nt) {
      v16bf vf;
      const __bf16* p = vp + (size_t)(nt*16 + l16) * Ss + kvb + half*8;
      ((v8bf*)&vf)[0] = *(const v8bf*)p;
      ((v8bf*)&vf)[1] = *(const v8bf*)(p + 16);
      oacc[nt] = WMMA_BF16(pf, vf, oacc[nt]);
    }
  }

  const int b = bh >> 4, h = bh & (Hh - 1);
  #pragma unroll
  for (int r = 0; r < 8; ++r) {
    const float rinv = 1.0f / lrow[r];
    const int qrow = qbase + r + 8*half;
    #pragma unroll
    for (int nt = 0; nt < 4; ++nt)
      X[((size_t)b*Ss + qrow)*Ee + h*Dh + nt*16 + l16] = (__bf16)(oacc[nt][r] * rinv);
  }
}

extern "C" void kernel_launch(void* const* d_in, const int* in_sizes, int n_in,
                              void* d_out, int out_size, void* d_ws, size_t ws_size,
                              hipStream_t stream)
{
  (void)in_sizes; (void)n_in; (void)out_size; (void)ws_size;
  const float* v  = (const float*)d_in[0];
  const float* k  = (const float*)d_in[1];
  const float* q  = (const float*)d_in[2];
  // d_in[3] = causal tril mask -> applied analytically inside attn_kernel
  const float* wq = (const float*)d_in[4];
  const float* bq = (const float*)d_in[5];
  const float* wk = (const float*)d_in[6];
  const float* bk = (const float*)d_in[7];
  const float* wv = (const float*)d_in[8];
  const float* bv = (const float*)d_in[9];
  const float* wo = (const float*)d_in[10];
  const float* bo = (const float*)d_in[11];

  const size_t NEL  = (size_t)Bb * Hh * Ss * Dh;  // 8,388,608
  const size_t WEL  = (size_t)Ee * Ee;            // 1,048,576
  __bf16* qb  = (__bf16*)d_ws;      // bf16 copies of activations
  __bf16* kb  = qb  + NEL;
  __bf16* vb  = kb  + NEL;
  __bf16* wqb = vb  + NEL;          // bf16 copies of weights
  __bf16* wkb = wqb + WEL;
  __bf16* wvb = wkb + WEL;
  __bf16* wob = wvb + WEL;
  __bf16* Qh  = wob + WEL;          // projected per-head tensors
  __bf16* Kh  = Qh  + NEL;
  __bf16* Vh  = Kh  + NEL;
  __bf16* X   = Vh  + NEL;          // total ws: ~125 MiB

  dim3 blk(256);
  const int gA = (int)(NEL / 8 / 256);   // 4096 blocks, exact
  const int gW = (int)(WEL / 8 / 256);   // 512 blocks, exact
  cvt_f32_bf16_kernel<<<gA, blk, 0, stream>>>(q,  qb,  (int)NEL);
  cvt_f32_bf16_kernel<<<gA, blk, 0, stream>>>(k,  kb,  (int)NEL);
  cvt_f32_bf16_kernel<<<gA, blk, 0, stream>>>(v,  vb,  (int)NEL);
  cvt_f32_bf16_kernel<<<gW, blk, 0, stream>>>(wq, wqb, (int)WEL);
  cvt_f32_bf16_kernel<<<gW, blk, 0, stream>>>(wk, wkb, (int)WEL);
  cvt_f32_bf16_kernel<<<gW, blk, 0, stream>>>(wv, wvb, (int)WEL);
  cvt_f32_bf16_kernel<<<gW, blk, 0, stream>>>(wo, wob, (int)WEL);

  dim3 gg(Mm / 128, Ee / 128);  // 64 x 8
  gemm_bf16_kernel<0><<<gg, blk, 0, stream>>>(qb, wqb, bq, Qh);
  gemm_bf16_kernel<0><<<gg, blk, 0, stream>>>(kb, wkb, bk, Kh);
  gemm_bf16_kernel<1><<<gg, blk, 0, stream>>>(vb, wvb, bv, Vh);
  attn_kernel<<<dim3(Ss / 128, Bb * Hh), blk, 0, stream>>>(Qh, Kh, Vh, X);
  gemm_bf16_kernel<2><<<gg, blk, 0, stream>>>(X, wob, bo, (float*)d_out);
}
